// hd_rp_encoder_73581379715545
// MI455X (gfx1250) — matrix operand order repaired and checked by reading, not compiled, not run on hardware
//
#include <hip/hip_runtime.h>
#include <hip/hip_bf16.h>

// ---------------------------------------------------------------------------
// HD random-projection encoder == GEMM [4096,1024] x [1024,10000] -> fp32.
// MI455X (gfx1250, wave32) plan:
//   * operands converted to f16 (W is exactly +/-1 -> exact; x err ~2^-11/elem,
//     f32 WMMA accumulation keeps total error ~3e-4 relative)
//   * operands pre-swizzled in d_ws into V_WMMA_F32_16X16X32_F16 per-lane
//     fragment layout -> hot loop is pure b128 traffic + WMMA
//   * block tile 64(M) x 256(N), 8 waves (2x4), wave tile 32x64, K step 32
//   * global -> LDS via GLOBAL_LOAD_ASYNC_TO_LDS_B128 (ASYNCcnt), TRIPLE
//     buffered: each wave has 5 async loads per tile, s_wait_asynccnt 5
//     retires exactly the oldest tile while the next tile stays in flight
//   * #pragma unroll 1 + rotating buffer index keeps the 64 accumulator
//     VGPRs in one fixed set (kills the v_mov_b64 churn seen in round 3)
//   * non-temporal output stores: the 164 MB output stream must not evict the
//     L2-resident f16 operands (~29 MB inside the 192 MB L2)
// ---------------------------------------------------------------------------

typedef __attribute__((ext_vector_type(16))) _Float16 v16h;
typedef __attribute__((ext_vector_type(8)))  _Float16 half8;
typedef __attribute__((ext_vector_type(8)))  float    v8f;
typedef __attribute__((ext_vector_type(4)))  int      v4i;

#define SIZE_IN 1024
#define D_OUT   10000
#define BATCH   4096
#define NPAD    10240          // 40 * 256 (pad N so GEMM loads need no masking)
#define KTILES  32             // 1024 / 32
#define A_SW_HALVES (256 * KTILES * 512)   //  4,194,304 halves ( 8 MiB)
#define B_SW_HALVES (640 * KTILES * 512)   // 10,485,760 halves (20 MiB)

#if defined(__has_builtin)
#  if __has_builtin(__builtin_amdgcn_global_load_async_to_lds_b128) && \
      __has_builtin(__builtin_amdgcn_s_wait_asynccnt)
#    define USE_ASYNC_LDS 1
#  endif
#endif
#ifndef USE_ASYNC_LDS
#  define USE_ASYNC_LDS 0
#endif

// Per-lane K index for a 16-bit A fragment (ISA 7.12.2, 16-bit A 16x32):
// lanes 0-15: VGPR0-3 -> K 0..7, VGPR4-7 -> K 16..23 ; lanes 16-31: +8
__device__ __forceinline__ int kmapA(int lane, int j) {
    return ((j < 8) ? j : (j + 8)) + ((lane & 16) ? 8 : 0);
}
// Per-lane K index for a 16-bit B fragment (same K order as A):
// lanes 0-15: K 0..7, 16..23 ; lanes 16-31: +8
__device__ __forceinline__ int kmapB(int lane, int j) {
    return ((j < 8) ? j : (j + 8)) + ((lane & 16) ? 8 : 0);
}

// x[4096,1024] fp32 -> A_sw fragments [mt(256)][kt(32)][512 lane-ordered halves]
__global__ void pack_a_kernel(const float* __restrict__ x,
                              _Float16* __restrict__ A_sw) {
    int tid  = blockIdx.x * blockDim.x + threadIdx.x;   // 0 .. A_SW_HALVES-1
    int idx  = tid & 511;
    int frag = tid >> 9;
    int lane = idx >> 4;
    int j    = idx & 15;
    int mt   = frag >> 5;        // frag / KTILES
    int kt   = frag & 31;
    int m    = mt * 16 + (lane & 15);
    int k    = kt * 32 + kmapA(lane, j);
    A_sw[tid] = (_Float16)x[m * SIZE_IN + k];
}

// W[1024,10000] fp32 -> B_sw fragments [nt(640)][kt(32)][512 lane-ordered halves]
// (implicitly transposes to N-major and zero-pads N to 10240)
__global__ void pack_b_kernel(const float* __restrict__ w,
                              _Float16* __restrict__ B_sw) {
    int tid  = blockIdx.x * blockDim.x + threadIdx.x;   // 0 .. B_SW_HALVES-1
    int idx  = tid & 511;
    int frag = tid >> 9;
    int lane = idx >> 4;
    int j    = idx & 15;
    int nt   = frag >> 5;
    int kt   = frag & 31;
    int n    = nt * 16 + (lane & 15);
    int k    = kt * 32 + kmapB(lane, j);
    float v  = (n < D_OUT) ? w[k * D_OUT + n] : 0.0f;
    B_sw[tid] = (_Float16)v;
}

__global__ __launch_bounds__(256)
void gemm_wmma_kernel(const half8* __restrict__ Ag,   // swizzled x (f16)
                      const half8* __restrict__ Bg,   // swizzled Wt (f16)
                      float* __restrict__ out) {
#if USE_ASYNC_LDS
    __shared__ half8 As8[3][4 * 64];    // 12 KiB (triple-buffered A tile)
    __shared__ half8 Bs8[3][16 * 64];   // 48 KiB (triple-buffered B tile)
#else
    __shared__ half8 As8[1][4 * 64];
    __shared__ half8 Bs8[1][16 * 64];
#endif

    const int t    = threadIdx.x;
    const int lane = t & 31;
    const int wave = t >> 5;
    const int wm   = wave & 1;      // 0..1 : which 32-row half of the block
    const int wn   = wave >> 1;     // 0..3 : which 64-col quarter

    const half8* Abase = Ag + (size_t)(blockIdx.y * 4)  * KTILES * 64;
    const half8* Bbase = Bg + (size_t)(blockIdx.x * 16) * KTILES * 64;

    // copy assignment: A = 256 chunks (1/thread), B = 1024 chunks (4/thread)
    const int ai  = t >> 6,          ao  = t & 63;
    const int bi0 = (t      ) >> 6,  bo0 = (t      ) & 63;
    const int bi1 = (t + 256) >> 6,  bo1 = (t + 256) & 63;
    const int bi2 = (t + 512) >> 6,  bo2 = (t + 512) & 63;
    const int bi3 = (t + 768) >> 6,  bo3 = (t + 768) & 63;

    v8f acc[2][4] = {};
    union Frag { v16h v; half8 h[2]; };

    // ---- per-K-step compute: 2 A frags, 4 B frags, 8 WMMAs ----
    auto compute = [&](const half8* __restrict__ A,
                       const half8* __restrict__ B) {
        Frag a0, a1, b0, b1, b2, b3;
        const int la = lane * 2;
        a0.h[0] = A[(wm * 2 + 0) * 64 + la];
        a0.h[1] = A[(wm * 2 + 0) * 64 + la + 1];
        a1.h[0] = A[(wm * 2 + 1) * 64 + la];
        a1.h[1] = A[(wm * 2 + 1) * 64 + la + 1];
        b0.h[0] = B[(wn * 4 + 0) * 64 + la];
        b0.h[1] = B[(wn * 4 + 0) * 64 + la + 1];
        b1.h[0] = B[(wn * 4 + 1) * 64 + la];
        b1.h[1] = B[(wn * 4 + 1) * 64 + la + 1];
        b2.h[0] = B[(wn * 4 + 2) * 64 + la];
        b2.h[1] = B[(wn * 4 + 2) * 64 + la + 1];
        b3.h[0] = B[(wn * 4 + 3) * 64 + la];
        b3.h[1] = B[(wn * 4 + 3) * 64 + la + 1];
        acc[0][0] = __builtin_amdgcn_wmma_f32_16x16x32_f16(false, a0.v, false, b0.v, (short)0, acc[0][0], false, false);
        acc[0][1] = __builtin_amdgcn_wmma_f32_16x16x32_f16(false, a0.v, false, b1.v, (short)0, acc[0][1], false, false);
        acc[0][2] = __builtin_amdgcn_wmma_f32_16x16x32_f16(false, a0.v, false, b2.v, (short)0, acc[0][2], false, false);
        acc[0][3] = __builtin_amdgcn_wmma_f32_16x16x32_f16(false, a0.v, false, b3.v, (short)0, acc[0][3], false, false);
        acc[1][0] = __builtin_amdgcn_wmma_f32_16x16x32_f16(false, a1.v, false, b0.v, (short)0, acc[1][0], false, false);
        acc[1][1] = __builtin_amdgcn_wmma_f32_16x16x32_f16(false, a1.v, false, b1.v, (short)0, acc[1][1], false, false);
        acc[1][2] = __builtin_amdgcn_wmma_f32_16x16x32_f16(false, a1.v, false, b2.v, (short)0, acc[1][2], false, false);
        acc[1][3] = __builtin_amdgcn_wmma_f32_16x16x32_f16(false, a1.v, false, b3.v, (short)0, acc[1][3], false, false);
    };

#if USE_ASYNC_LDS
    // ---- async direct-to-LDS pipeline (ASYNCcnt), triple-buffered ----
    // builtin signature: (int4 AS(1)*, int4 AS(3)*, imm int offset, imm int cpol)
    typedef __attribute__((address_space(1))) v4i as1_v4i;
    typedef __attribute__((address_space(3))) v4i as3_v4i;

    // running per-thread global pointers; consecutive K-tiles are +64 half8
    const half8* pA  = Abase + ai  * KTILES * 64 + ao;
    const half8* pB0 = Bbase + bi0 * KTILES * 64 + bo0;
    const half8* pB1 = Bbase + bi1 * KTILES * 64 + bo1;
    const half8* pB2 = Bbase + bi2 * KTILES * 64 + bo2;
    const half8* pB3 = Bbase + bi3 * KTILES * 64 + bo3;

    auto issue = [&](int bufn) {   // 5 async loads per wave per tile
        __builtin_amdgcn_global_load_async_to_lds_b128(
            (as1_v4i*)pA,  (as3_v4i*)&As8[bufn][t],       0, 0);
        __builtin_amdgcn_global_load_async_to_lds_b128(
            (as1_v4i*)pB0, (as3_v4i*)&Bs8[bufn][t],       0, 0);
        __builtin_amdgcn_global_load_async_to_lds_b128(
            (as1_v4i*)pB1, (as3_v4i*)&Bs8[bufn][t + 256], 0, 0);
        __builtin_amdgcn_global_load_async_to_lds_b128(
            (as1_v4i*)pB2, (as3_v4i*)&Bs8[bufn][t + 512], 0, 0);
        __builtin_amdgcn_global_load_async_to_lds_b128(
            (as1_v4i*)pB3, (as3_v4i*)&Bs8[bufn][t + 768], 0, 0);
        pA += 64; pB0 += 64; pB1 += 64; pB2 += 64; pB3 += 64;
    };

    issue(0);                       // tile 0 -> buf 0
    issue(1);                       // tile 1 -> buf 1
    int bc = 0;                     // buffer holding tile kt
#pragma unroll 1
    for (int kt = 0; kt < KTILES - 1; ++kt) {
        __builtin_amdgcn_s_wait_asynccnt(5);  // oldest tile (kt) resident;
                                              // tile kt+1's 5 loads in flight
        __syncthreads();                      // all waves: tile kt landed and
                                              // everyone finished buf reuse
        if (kt + 2 < KTILES) {
            int bI = bc - 1; if (bI < 0) bI += 3;   // (bc + 2) % 3
            issue(bI);                              // tile kt+2
        }
        compute(&As8[bc][0], &Bs8[bc][0]);
        ++bc; if (bc == 3) bc = 0;
    }
    __builtin_amdgcn_s_wait_asynccnt(0);      // last tile resident
    __syncthreads();
    compute(&As8[bc][0], &Bs8[bc][0]);
#else
    // ---- fallback: global -> VGPR -> ds_store staging (register dbl-buffer)
    half8 ra, rb0, rb1, rb2, rb3;
    auto loadK = [&](int kt) {
        ra  = Abase[(ai  * KTILES + kt) * 64 + ao ];
        rb0 = Bbase[(bi0 * KTILES + kt) * 64 + bo0];
        rb1 = Bbase[(bi1 * KTILES + kt) * 64 + bo1];
        rb2 = Bbase[(bi2 * KTILES + kt) * 64 + bo2];
        rb3 = Bbase[(bi3 * KTILES + kt) * 64 + bo3];
    };
    loadK(0);
    for (int kt = 0; kt < KTILES; ++kt) {
        __syncthreads();
        As8[0][t]       = ra;
        Bs8[0][t]       = rb0;
        Bs8[0][t + 256] = rb1;
        Bs8[0][t + 512] = rb2;
        Bs8[0][t + 768] = rb3;
        __syncthreads();
        if (kt + 1 < KTILES) loadK(kt + 1);
        compute(&As8[0][0], &Bs8[0][0]);
    }
#endif

    // C/D layout (ISA 7.12.2): VGPR v, lanes 0-15 -> row v, N=lane;
    //                          lanes 16-31 -> row v+8, N=lane-16.
    const int mrow0 = blockIdx.y * 64 + wm * 32;
    const int ncol0 = blockIdx.x * 256 + wn * 64;
    const int nIn   = lane & 15;
    const int mHalf = (lane >> 4) * 8;
    for (int i = 0; i < 2; ++i) {
        for (int j = 0; j < 4; ++j) {
            int col = ncol0 + j * 16 + nIn;
            if (col < D_OUT) {
                for (int v = 0; v < 8; ++v) {
                    int row = mrow0 + i * 16 + mHalf + v;
                    __builtin_nontemporal_store(acc[i][j][v],
                                                out + (size_t)row * D_OUT + col);
                }
            }
        }
    }
}

extern "C" void kernel_launch(void* const* d_in, const int* in_sizes, int n_in,
                              void* d_out, int out_size, void* d_ws, size_t ws_size,
                              hipStream_t stream) {
    const float* x = (const float*)d_in[0];   // [4096, 32, 32] -> [4096,1024]
    const float* w = (const float*)d_in[1];   // [1024, 10000]
    float* out = (float*)d_out;               // [4096, 10000]

    _Float16* A_sw = (_Float16*)d_ws;                                  // 8 MiB
    _Float16* B_sw = (_Float16*)((char*)d_ws + (size_t)A_SW_HALVES*2); // 20 MiB

    pack_a_kernel<<<A_SW_HALVES / 256, 256, 0, stream>>>(x, A_sw);
    pack_b_kernel<<<B_SW_HALVES / 256, 256, 0, stream>>>(w, B_sw);

    dim3 grid(NPAD / 256, BATCH / 64);        // (40, 64)
    gemm_wmma_kernel<<<grid, 256, 0, stream>>>((const half8*)A_sw,
                                               (const half8*)B_sw, out);
}
